// CasualSelfAttention_25615184953949
// MI455X (gfx1250) — matrix operand order repaired and checked by reading, not compiled, not run on hardware
//
#include <hip/hip_runtime.h>
#include <stdint.h>

// ---------------------------------------------------------------------------
// Causal self attention, MI455X (gfx1250). All GEMM stages on
// v_wmma_f32_16x16x32_bf16 (fp32 accum), softmax in fp32 VALU.
// GEMM tiles staged via GLOBAL_LOAD_ASYNC_TO_LDS_B128 (ASYNCcnt) with
// double-buffered LDS when the toolchain exposes the builtins.
// B=2, T=2048, C=1024, H=16, Dh=64.
// ---------------------------------------------------------------------------

#define T_SEQ 2048
#define C_DIM 1024
#define N_H   16
#define DH    64

#if defined(__has_builtin)
#if __has_builtin(__builtin_amdgcn_global_load_async_to_lds_b128) && \
    __has_builtin(__builtin_amdgcn_s_wait_asynccnt)
#define USE_ASYNC_COPY 1
#endif
#endif

typedef __attribute__((ext_vector_type(16))) __bf16 v16bf;
typedef __attribute__((ext_vector_type(8)))  float  v8f;
typedef __attribute__((ext_vector_type(4)))  int    v4i;

union Frag16 { uint4 u[2]; v16bf v; };

#ifdef USE_ASYNC_COPY
typedef __attribute__((address_space(1))) v4i* glb_v4i_p;
typedef __attribute__((address_space(3))) v4i* lds_v4i_p;
__device__ __forceinline__ void async_copy16(const void* g, void* l) {
  __builtin_amdgcn_global_load_async_to_lds_b128(
      (glb_v4i_p)(g), (lds_v4i_p)(l), /*imm offset=*/0, /*cpol=*/0);
}
#endif

__device__ __forceinline__ uint16_t f2bf(float x) {
  union { float f; uint32_t u; } a; a.f = x;
  uint32_t u = a.u;
  uint32_t r = u + 0x7FFFu + ((u >> 16) & 1u);   // round-to-nearest-even
  return (uint16_t)(r >> 16);
}

__device__ __forceinline__ v8f wmma_bf16(v16bf a, v16bf b, v8f c) {
  return __builtin_amdgcn_wmma_f32_16x16x32_bf16(
      /*neg_a=*/false, a, /*neg_b=*/false, b,
      /*c_mod=*/(short)0, c, /*reuse_a=*/false, /*reuse_b=*/false);
}

// A fragment (16x32 bf16). Lane row = lane&15 for both halves; half = lane>>4
// selects K in [16h, 16h+16) ascending (contiguous, matching B fragment).
__device__ __forceinline__ v16bf load_a_frag(const uint16_t* row, int hgrp) {
  Frag16 f;
  f.u[0] = *(const uint4*)(row + hgrp * 16);
  f.u[1] = *(const uint4*)(row + hgrp * 16 + 8);
  return f.v;
}
// B fragment (32x16 bf16). Lane col = lane&15; lane group g = lane>>4 holds
// K in [16g, 16g+16) ascending -> 16 contiguous bf16 from a K-major row.
__device__ __forceinline__ v16bf load_b_frag(const uint16_t* p) {
  Frag16 f;
  f.u[0] = *(const uint4*)(p);
  f.u[1] = *(const uint4*)(p + 8);
  return f.v;
}

// ---------------------------------------------------------------------------
__global__ __launch_bounds__(256) void cast_f32_to_bf16(const float* __restrict__ in,
                                                        uint16_t* __restrict__ out,
                                                        int n) {
  for (int i = blockIdx.x * blockDim.x + threadIdx.x; i < n;
       i += gridDim.x * blockDim.x)
    out[i] = f2bf(in[i]);
}

// ---------------------------------------------------------------------------
// Shared GEMM mainloop body: block tile 128x128, waves 4(M) x 2(N), each wave
// 32x64 (2x4 WMMA subtiles), K steps of 32 over C_DIM.
// acc must be v8f[2][4]. Defined as a macro so both kernels share it.
#ifdef USE_ASYNC_COPY

#define GEMM_MAINLOOP(SRC_A, SRC_B)                                            \
  {                                                                            \
    auto issue = [&](int it, int buf) {                                        \
      const uint16_t* ga =                                                     \
          (SRC_A) + (size_t)(m0 + lrow) * C_DIM + it * 32 + lseg * 16;         \
      const uint16_t* gb =                                                     \
          (SRC_B) + (size_t)(n0 + lrow) * C_DIM + it * 32 + lseg * 16;         \
      async_copy16(ga,     &As[buf][lrow][lseg * 16]);                         \
      async_copy16(ga + 8, &As[buf][lrow][lseg * 16 + 8]);                     \
      async_copy16(gb,     &Bs[buf][lrow][lseg * 16]);                         \
      async_copy16(gb + 8, &Bs[buf][lrow][lseg * 16 + 8]);                     \
    };                                                                         \
    const int NT = C_DIM / 32;                                                 \
    issue(0, 0);                                                               \
    for (int it = 0; it < NT; ++it) {                                          \
      const int cur = it & 1;                                                  \
      if (it + 1 < NT) {                                                       \
        issue(it + 1, cur ^ 1);            /* overlap copy with WMMA */        \
        __builtin_amdgcn_s_wait_asynccnt(4);  /* tile it complete */           \
      } else {                                                                 \
        __builtin_amdgcn_s_wait_asynccnt(0);  /* drain */                      \
      }                                                                        \
      __syncthreads();                                                         \
      v16bf aF[2], bF[4];                                                      \
      _Pragma("unroll")                                                        \
      for (int mi = 0; mi < 2; ++mi)                                           \
        aF[mi] = load_a_frag(&As[cur][waveM * 32 + mi * 16 + col][0], hgrp);   \
      _Pragma("unroll")                                                        \
      for (int ni = 0; ni < 4; ++ni)                                           \
        bF[ni] = load_b_frag(&Bs[cur][waveN * 64 + ni * 16 + col][hgrp * 16]); \
      _Pragma("unroll")                                                        \
      for (int mi = 0; mi < 2; ++mi)                                           \
        _Pragma("unroll")                                                      \
        for (int ni = 0; ni < 4; ++ni)                                         \
          acc[mi][ni] = wmma_bf16(aF[mi], bF[ni], acc[mi][ni]);                \
      __syncthreads();                                                         \
    }                                                                          \
  }

#define GEMM_LDS_DECL                                                          \
  __shared__ __align__(16) uint16_t As[2][128][32];                            \
  __shared__ __align__(16) uint16_t Bs[2][128][32];

#else  // !USE_ASYNC_COPY — proven VGPR-staged fallback

#define GEMM_MAINLOOP(SRC_A, SRC_B)                                            \
  for (int k0 = 0; k0 < C_DIM; k0 += 32) {                                     \
    const uint16_t* ga = (SRC_A) + (size_t)(m0 + lrow) * C_DIM + k0 + lseg*16; \
    const uint16_t* gb = (SRC_B) + (size_t)(n0 + lrow) * C_DIM + k0 + lseg*16; \
    *(uint4*)(&As[0][lrow][lseg * 16])     = *(const uint4*)(ga);              \
    *(uint4*)(&As[0][lrow][lseg * 16 + 8]) = *(const uint4*)(ga + 8);          \
    *(uint4*)(&Bs[0][lrow][lseg * 16])     = *(const uint4*)(gb);              \
    *(uint4*)(&Bs[0][lrow][lseg * 16 + 8]) = *(const uint4*)(gb + 8);          \
    if (k0 + 32 < C_DIM) {                                                     \
      __builtin_prefetch(ga + 32, 0, 0);                                       \
      __builtin_prefetch(gb + 32, 0, 0);                                       \
    }                                                                          \
    __syncthreads();                                                           \
    v16bf aF[2], bF[4];                                                        \
    _Pragma("unroll")                                                          \
    for (int mi = 0; mi < 2; ++mi)                                             \
      aF[mi] = load_a_frag(&As[0][waveM * 32 + mi * 16 + col][0], hgrp);       \
    _Pragma("unroll")                                                          \
    for (int ni = 0; ni < 4; ++ni)                                             \
      bF[ni] = load_b_frag(&Bs[0][waveN * 64 + ni * 16 + col][hgrp * 16]);     \
    _Pragma("unroll")                                                          \
    for (int mi = 0; mi < 2; ++mi)                                             \
      _Pragma("unroll")                                                        \
      for (int ni = 0; ni < 4; ++ni)                                           \
        acc[mi][ni] = wmma_bf16(aF[mi], bF[ni], acc[mi][ni]);                  \
    __syncthreads();                                                           \
  }

#define GEMM_LDS_DECL                                                          \
  __shared__ __align__(16) uint16_t As[1][128][32];                            \
  __shared__ __align__(16) uint16_t Bs[1][128][32];

#endif  // USE_ASYNC_COPY

// ---------------------------------------------------------------------------
// QKV GEMM: qkv[m, j] = sum_c x[m,c] * w_attn[j,c] + b_attn[j]
// M = B*T = 4096, N = 3C = 3072, K = 1024. Epilogue scatters into q,k
// ([B,H,T,Dh]) and v transposed ([B,H,Dh,T]).
__global__ __launch_bounds__(256) void qkv_gemm(const uint16_t* __restrict__ X,
                                                const uint16_t* __restrict__ W,
                                                const float* __restrict__ bias,
                                                uint16_t* __restrict__ Qo,
                                                uint16_t* __restrict__ Ko,
                                                uint16_t* __restrict__ Vo) {
  GEMM_LDS_DECL

  const int tid   = threadIdx.x;
  const int lane  = tid & 31;
  const int wave  = tid >> 5;
  const int waveM = wave & 3;      // 4 waves in M
  const int waveN = wave >> 2;     // 2 waves in N
  const int m0 = blockIdx.y * 128;
  const int n0 = blockIdx.x * 128;
  const int lrow = tid >> 1, lseg = tid & 1;   // tile loader mapping
  const int col  = lane & 15, hgrp = lane >> 4;

  v8f acc[2][4];
  v8f zero = {};
#pragma unroll
  for (int i = 0; i < 2; ++i)
#pragma unroll
    for (int j = 0; j < 4; ++j) acc[i][j] = zero;

  GEMM_MAINLOOP(X, W)

  // Epilogue: C tile layout -> row M = v + 8*hgrp, col N = lane&15.
#pragma unroll
  for (int mi = 0; mi < 2; ++mi)
#pragma unroll
    for (int ni = 0; ni < 4; ++ni) {
      const int j = n0 + waveN * 64 + ni * 16 + col;
      const float bj = bias[j];
#pragma unroll
      for (int v = 0; v < 8; ++v) {
        const int m = m0 + waveM * 32 + mi * 16 + v + 8 * hgrp;
        const int b = m >> 11, t = m & (T_SEQ - 1);
        const uint16_t val = f2bf(acc[mi][ni][v] + bj);
        if (j < C_DIM) {
          const int hh = j >> 6, d = j & 63;
          Qo[(((size_t)b * N_H + hh) * T_SEQ + t) * DH + d] = val;
        } else if (j < 2 * C_DIM) {
          const int jj = j - C_DIM, hh = jj >> 6, d = jj & 63;
          Ko[(((size_t)b * N_H + hh) * T_SEQ + t) * DH + d] = val;
        } else {
          const int jj = j - 2 * C_DIM, hh = jj >> 6, d = jj & 63;
          Vo[(((size_t)b * N_H + hh) * DH + d) * T_SEQ + t] = val;   // V^T
        }
      }
    }
}

// ---------------------------------------------------------------------------
// Flash attention: one block per (b*H+h, q-block of 128). 8 waves, 16 q rows
// per wave. Online softmax, 32-wide k chunks.
__global__ __launch_bounds__(256) void flash_attn(const uint16_t* __restrict__ Q,
                                                  const uint16_t* __restrict__ K,
                                                  const uint16_t* __restrict__ V,
                                                  uint16_t* __restrict__ AO) {
  __shared__ __align__(16) uint16_t pTile[8][16][32];

  const int tid  = threadIdx.x;
  const int lane = tid & 31;
  const int wave = tid >> 5;
  const int col  = lane & 15, hgrp = lane >> 4;
  const int bh = blockIdx.x;                    // b*16 + h
  const int b  = bh >> 4, h = bh & 15;
  const int qbase = blockIdx.y * 128 + wave * 16;
  const size_t baseTD = (size_t)bh * T_SEQ * DH;   // [T][Dh] base for Q,K
  const size_t baseDT = (size_t)bh * DH * T_SEQ;   // [Dh][T] base for V^T
  const float NEG_INF = -__builtin_inff();

  // Q fragments (Dh=64 -> two K steps of 32)
  v16bf aQ[2];
  {
    const uint16_t* qrow = Q + baseTD + (size_t)(qbase + col) * DH;
#pragma unroll
    for (int s = 0; s < 2; ++s) {
      Frag16 f;
      f.u[0] = *(const uint4*)(qrow + s * 32 + hgrp * 16);
      f.u[1] = *(const uint4*)(qrow + s * 32 + hgrp * 16 + 8);
      aQ[s] = f.v;
    }
  }

  float mrun[8], lrun[8];
  v8f accO[4];
  v8f zero = {};
#pragma unroll
  for (int v = 0; v < 8; ++v) { mrun[v] = NEG_INF; lrun[v] = 0.f; }
#pragma unroll
  for (int d = 0; d < 4; ++d) accO[d] = zero;

  const int nchunks = (blockIdx.y + 1) * 4;      // causal bound, 32-wide chunks
  for (int c = 0; c < nchunks; ++c) {
    const int kb = c * 32;
    // S = Q K^T : two 16x16 column subtiles
    v8f S[2];
    S[0] = zero; S[1] = zero;
#pragma unroll
    for (int ni = 0; ni < 2; ++ni)
#pragma unroll
      for (int s = 0; s < 2; ++s) {
        const uint16_t* krow =
            K + baseTD + (size_t)(kb + ni * 16 + col) * DH + s * 32 + hgrp * 16;
        S[ni] = wmma_bf16(aQ[s], load_b_frag(krow), S[ni]);
      }

    // online softmax per row (row lives in one 16-lane half-group)
#pragma unroll
    for (int v = 0; v < 8; ++v) {
      const int qg = qbase + v + 8 * hgrp;
      float s0 = S[0][v] * 0.125f;               // 1/sqrt(Dh)
      float s1 = S[1][v] * 0.125f;
      if (kb + col > qg)      s0 = NEG_INF;
      if (kb + 16 + col > qg) s1 = NEG_INF;
      float t = fmaxf(s0, s1);
      t = fmaxf(t, __shfl_xor(t, 1, 32));
      t = fmaxf(t, __shfl_xor(t, 2, 32));
      t = fmaxf(t, __shfl_xor(t, 4, 32));
      t = fmaxf(t, __shfl_xor(t, 8, 32));
      const float mnew = fmaxf(mrun[v], t);
      const float fac  = __expf(mrun[v] - mnew);
      const float p0 = __expf(s0 - mnew);
      const float p1 = __expf(s1 - mnew);
      float rs = p0 + p1;
      rs += __shfl_xor(rs, 1, 32);
      rs += __shfl_xor(rs, 2, 32);
      rs += __shfl_xor(rs, 4, 32);
      rs += __shfl_xor(rs, 8, 32);
      lrun[v] = lrun[v] * fac + rs;
      mrun[v] = mnew;
#pragma unroll
      for (int d = 0; d < 4; ++d) accO[d][v] *= fac;
      pTile[wave][v + 8 * hgrp][col]      = f2bf(p0);
      pTile[wave][v + 8 * hgrp][16 + col] = f2bf(p1);
    }

    // O += P V   (A = P from per-wave LDS tile, B = V^T rows, contiguous K=t)
    Frag16 fp;
    fp.u[0] = *(const uint4*)(&pTile[wave][col][hgrp * 16]);
    fp.u[1] = *(const uint4*)(&pTile[wave][col][hgrp * 16 + 8]);
#pragma unroll
    for (int d = 0; d < 4; ++d) {
      const uint16_t* vrow =
          V + baseDT + (size_t)(d * 16 + col) * T_SEQ + kb + hgrp * 16;
      accO[d] = wmma_bf16(fp.v, load_b_frag(vrow), accO[d]);
    }
  }

  // normalize + write attention output [B,T,C] (c = h*64 + d) as bf16
#pragma unroll
  for (int d = 0; d < 4; ++d)
#pragma unroll
    for (int v = 0; v < 8; ++v) {
      const int qg = qbase + v + 8 * hgrp;
      const float o = accO[d][v] / lrun[v];
      AO[((size_t)b * T_SEQ + qg) * C_DIM + h * DH + d * 16 + col] = f2bf(o);
    }
}

// ---------------------------------------------------------------------------
// Output projection: out[m, j] = sum_c ao[m,c] * w_proj[j,c] + b_proj[j]
// M = 4096, N = 1024, K = 1024. fp32 output directly to d_out.
__global__ __launch_bounds__(256) void proj_gemm(const uint16_t* __restrict__ A,
                                                 const uint16_t* __restrict__ W,
                                                 const float* __restrict__ bias,
                                                 float* __restrict__ Out) {
  GEMM_LDS_DECL

  const int tid   = threadIdx.x;
  const int lane  = tid & 31;
  const int wave  = tid >> 5;
  const int waveM = wave & 3;
  const int waveN = wave >> 2;
  const int m0 = blockIdx.y * 128;
  const int n0 = blockIdx.x * 128;
  const int lrow = tid >> 1, lseg = tid & 1;
  const int col  = lane & 15, hgrp = lane >> 4;

  v8f acc[2][4];
  v8f zero = {};
#pragma unroll
  for (int i = 0; i < 2; ++i)
#pragma unroll
    for (int j = 0; j < 4; ++j) acc[i][j] = zero;

  GEMM_MAINLOOP(A, W)

#pragma unroll
  for (int mi = 0; mi < 2; ++mi)
#pragma unroll
    for (int ni = 0; ni < 4; ++ni) {
      const int j = n0 + waveN * 64 + ni * 16 + col;
      const float bj = bias[j];
#pragma unroll
      for (int v = 0; v < 8; ++v) {
        const int m = m0 + waveM * 32 + mi * 16 + v + 8 * hgrp;
        Out[(size_t)m * C_DIM + j] = acc[mi][ni][v] + bj;
      }
    }
}

// ---------------------------------------------------------------------------
extern "C" void kernel_launch(void* const* d_in, const int* in_sizes, int n_in,
                              void* d_out, int out_size, void* d_ws, size_t ws_size,
                              hipStream_t stream) {
  const float* x      = (const float*)d_in[0];   // [2,2048,1024]
  const float* w_attn = (const float*)d_in[1];   // [3072,1024]
  const float* b_attn = (const float*)d_in[2];   // [3072]
  const float* w_proj = (const float*)d_in[3];   // [1024,1024]
  const float* b_proj = (const float*)d_in[4];   // [1024]
  float* out = (float*)d_out;                    // [2,2048,1024]

  const size_t nX  = (size_t)2 * T_SEQ * C_DIM;      // 4194304
  const size_t nWA = (size_t)3 * C_DIM * C_DIM;      // 3145728
  const size_t nWP = (size_t)C_DIM * C_DIM;          // 1048576
  const size_t nQ  = (size_t)2 * N_H * T_SEQ * DH;   // 4194304

  uint16_t* xb  = (uint16_t*)d_ws;
  uint16_t* wab = xb + nX;
  uint16_t* wpb = wab + nWA;
  uint16_t* qw  = wpb + nWP;
  uint16_t* kw  = qw + nQ;
  uint16_t* vw  = kw + nQ;       // transposed [B,H,Dh,T]
  uint16_t* ao  = vw + nQ;       // [B,T,C] bf16

  cast_f32_to_bf16<<<(int)((nX + 255) / 256), 256, 0, stream>>>(x, xb, (int)nX);
  cast_f32_to_bf16<<<(int)((nWA + 255) / 256), 256, 0, stream>>>(w_attn, wab, (int)nWA);
  cast_f32_to_bf16<<<(int)((nWP + 255) / 256), 256, 0, stream>>>(w_proj, wpb, (int)nWP);

  qkv_gemm<<<dim3(24, 32), 256, 0, stream>>>(xb, wab, b_attn, qw, kw, vw);
  flash_attn<<<dim3(32, 16), 256, 0, stream>>>(qw, kw, vw, ao);
  proj_gemm<<<dim3(8, 32), 256, 0, stream>>>(ao, wpb, b_proj, out);
}
